// GatedXAttentionBlock_18932215841115
// MI455X (gfx1250) — matrix operand order repaired and checked by reading, not compiled, not run on hardware
//
#include <hip/hip_runtime.h>
#include <hip/hip_bf16.h>

// ---------------------------------------------------------------------------
// GatedXAttentionBlock for MI455X (gfx1250), wave32 + WMMA f16->f32.
//   - Weights are cast f32->f16 AND transposed once, so every GEMM operand is
//     fragment-contiguous: the GEMM is LDS-free / barrier-free, streaming
//     global_load_b128 -> v_wmma (operands stay resident in the 192MB L2).
//   - Flash attention stages K/V tiles into LDS once per block (8 waves share
//     them), via global_load_async_to_lds_b128 (+ s_wait_asynccnt) when
//     available; probs round-trip through per-wave LDS (C->A layout).
// ---------------------------------------------------------------------------

#ifndef USE_ASYNC_LDS
#define USE_ASYNC_LDS 1
#endif

typedef _Float16 half_t;
typedef __attribute__((ext_vector_type(16))) _Float16 v16h;
typedef __attribute__((ext_vector_type(8)))  float    v8f;

#define B_   2
#define TQ_  2048
#define TKV_ 2048
#define DIM_ 1024
#define HEADS_ 16
#define DHEAD_ 64
#define FFDIM_ 4096
#define MROWS_ (B_ * TQ_)

// -------------------------- WMMA / fragment helpers ------------------------

__device__ inline v8f wmma16(v16h a, v16h b, v8f c) {
  return __builtin_amdgcn_wmma_f32_16x16x32_f16(false, a, false, b, (short)0, c,
                                                false, false);
}

// A fragment (16x32 f16, MxK): element (m,k) = src[m*ld + k].
// lanes 0-15: m=lane, K {0..7,16..23}; lanes 16-31: K {8..15,24..31}.
__device__ inline v16h load_frag_a(const half_t* src, int ld) {
  const int lane = threadIdx.x & 31;
  const int m  = lane & 15;
  const int kb = (lane >> 4) * 8;
  const half_t* p = src + (size_t)m * ld + kb;
  union { v16h h; uint4 q[2]; } u;
  u.q[0] = *reinterpret_cast<const uint4*>(p);
  u.q[1] = *reinterpret_cast<const uint4*>(p + 16);
  return u.h;
}

// B fragment (32x16 f16, KxN): element (k,n) = src[n*ld + k] (K-contiguous).
// lanes 0-15: n=lane, K {0..7,16..23}; lanes 16-31: K {8..15,24..31}.
__device__ inline v16h load_frag_b(const half_t* src, int ld) {
  const int lane = threadIdx.x & 31;
  const int n  = lane & 15;
  const int kb = (lane >> 4) * 8;
  const half_t* p = src + (size_t)n * ld + kb;
  union { v16h h; uint4 q[2]; } u;
  u.q[0] = *reinterpret_cast<const uint4*>(p);
  u.q[1] = *reinterpret_cast<const uint4*>(p + 16);
  return u.h;
}

#if USE_ASYNC_LDS
// Per-lane 16B global->LDS async gather (GVS mode), tracked by ASYNCcnt.
__device__ inline void async_g2l_b128(void* lds_ptr, const void* gbase,
                                      unsigned byte_off) {
  unsigned l = (unsigned)(uintptr_t)lds_ptr;
  asm volatile("global_load_async_to_lds_b128 %0, %1, %2"
               :: "v"(l), "v"(byte_off), "s"(gbase) : "memory");
}
__device__ inline void async_wait0() {
  asm volatile("s_wait_asynccnt 0x0" ::: "memory");
}
#endif

// ----------------------- elementwise / prep kernels ------------------------

// f32 [R][C] -> f16 transposed [C][R]  (weights become fragment-contiguous)
__global__ void cast_transpose_kernel(const float* __restrict__ a,
                                      half_t* __restrict__ o, int R, int C) {
  int i = blockIdx.x * blockDim.x + threadIdx.x;
  if (i >= R * C) return;
  int r = i / C, c = i - r * C;
  o[(size_t)c * R + r] = (half_t)a[i];
}

// One block (256 threads) per 1024-wide row: layernorm + cast to f16.
__global__ __launch_bounds__(256)
void layernorm_cast_kernel(const float* __restrict__ x,
                           const float* __restrict__ gamma,
                           const float* __restrict__ beta,
                           half_t* __restrict__ out) {
  const int row = blockIdx.x;
  const float* xr = x + (size_t)row * DIM_;
  float vals[4], s = 0.f, s2 = 0.f;
#pragma unroll
  for (int i = 0; i < 4; ++i) {
    float v = xr[threadIdx.x + i * 256];
    vals[i] = v; s += v; s2 += v * v;
  }
  __shared__ float red0[256], red1[256];
  red0[threadIdx.x] = s; red1[threadIdx.x] = s2;
  __syncthreads();
  for (int off = 128; off > 0; off >>= 1) {
    if (threadIdx.x < off) {
      red0[threadIdx.x] += red0[threadIdx.x + off];
      red1[threadIdx.x] += red1[threadIdx.x + off];
    }
    __syncthreads();
  }
  const float mean = red0[0] * (1.0f / DIM_);
  const float var  = red1[0] * (1.0f / DIM_) - mean * mean;
  const float rstd = rsqrtf(var + 1e-5f);
  half_t* orow = out + (size_t)row * DIM_;
#pragma unroll
  for (int i = 0; i < 4; ++i) {
    int c = threadIdx.x + i * 256;
    orow[c] = (half_t)((vals[i] - mean) * rstd * gamma[c] + beta[c]);
  }
}

// v (cols 1024..2047 of kv_h [b*TKV][2048]) -> vt [((b*H+h)*64+d)][TKV]
__global__ void transpose_v_kernel(const half_t* __restrict__ kv,
                                   half_t* __restrict__ vt) {
  int idx = blockIdx.x * blockDim.x + threadIdx.x;  // (b, j, h, d)
  if (idx >= B_ * TKV_ * HEADS_ * DHEAD_) return;
  int d = idx & (DHEAD_ - 1);
  int h = (idx >> 6) & (HEADS_ - 1);
  int j = (idx >> 10) & (TKV_ - 1);
  int b = idx >> 21;
  half_t v = kv[(size_t)(b * TKV_ + j) * 2048 + 1024 + h * DHEAD_ + d];
  vt[((size_t)((b * HEADS_ + h) * DHEAD_ + d)) * TKV_ + j] = v;
}

// ------------------------- LDS-free WMMA GEMM ------------------------------
// C(MxN) = A(MxK f16 row-major) @ Bt^T, with Bt given TRANSPOSED [N][K].
// block = 256 thr = 8 waves (2x4); wave tile 64x32 = 4x2 wmma tiles.
// Fragments stream straight from global (L2-resident), no LDS, no barriers.
// mode 0: out_h=(f16)(acc*scale)  mode 1: out_h=(f16)gelu(acc)
// mode 2: out_f=acc*tanh(gate[0])+res

__global__ __launch_bounds__(256)
void gemm_wmma_kernel(const half_t* __restrict__ A, const half_t* __restrict__ Bt,
                      int K, int lda, int ldc, int mode, float scale,
                      const float* __restrict__ gate, const float* __restrict__ res,
                      half_t* __restrict__ out_h, float* __restrict__ out_f) {
  const int lane = threadIdx.x & 31;
  const int wave = threadIdx.x >> 5;
  const int waveM = (wave >> 2) * 64;
  const int waveN = (wave & 3) * 32;
  const int bm0 = blockIdx.y * 128;
  const int bn0 = blockIdx.x * 128;

  const half_t* arow[4];
  const half_t* brow[2];
#pragma unroll
  for (int mt = 0; mt < 4; ++mt)
    arow[mt] = A + (size_t)(bm0 + waveM + mt * 16) * lda;
#pragma unroll
  for (int nt = 0; nt < 2; ++nt)
    brow[nt] = Bt + (size_t)(bn0 + waveN + nt * 16) * K;

  v8f acc[4][2];
  v8f zero = {};
#pragma unroll
  for (int mt = 0; mt < 4; ++mt)
#pragma unroll
    for (int nt = 0; nt < 2; ++nt) acc[mt][nt] = zero;

  for (int k0 = 0; k0 < K; k0 += 32) {
    v16h af[4], bf[2];
#pragma unroll
    for (int mt = 0; mt < 4; ++mt) {
      af[mt] = load_frag_a(arow[mt] + k0, lda);
      __builtin_prefetch(arow[mt] + k0 + 32, 0, 0);
    }
#pragma unroll
    for (int nt = 0; nt < 2; ++nt) {
      bf[nt] = load_frag_b(brow[nt] + k0, K);
      __builtin_prefetch(brow[nt] + k0 + 32, 0, 0);
    }
#pragma unroll
    for (int mt = 0; mt < 4; ++mt)
#pragma unroll
      for (int nt = 0; nt < 2; ++nt)
        acc[mt][nt] = wmma16(af[mt], bf[nt], acc[mt][nt]);
  }

  // Epilogue: C/D element (vg, lane) -> row = vg + 8*(lane>=16), col = lane&15.
  const float gt = (mode == 2) ? tanhf(gate[0]) : 0.0f;
  const int hi = lane >> 4, nn = lane & 15;
#pragma unroll
  for (int mt = 0; mt < 4; ++mt)
#pragma unroll
    for (int nt = 0; nt < 2; ++nt)
#pragma unroll
      for (int vg = 0; vg < 8; ++vg) {
        int row = bm0 + waveM + mt * 16 + vg + 8 * hi;
        int col = bn0 + waveN + nt * 16 + nn;
        size_t idx = (size_t)row * ldc + col;
        float v = acc[mt][nt][vg];
        if (mode == 0) {
          out_h[idx] = (half_t)(v * scale);
        } else if (mode == 1) {
          float g = 0.5f * v *
              (1.0f + tanhf(0.7978845608028654f * (v + 0.044715f * v * v * v)));
          out_h[idx] = (half_t)g;
        } else {
          out_f[idx] = v * gt + res[idx];
        }
      }
}

// --------------------------- flash attention -------------------------------
// grid = B*H*(TQ/128); 256 thr = 8 waves, each owns 16 q rows of the block.
// Per 32-key step: block cooperatively stages K(32x64) and Vt(64x32) tiles
// into LDS (async-to-LDS when available), all 8 waves consume them; masked
// online softmax with 16-lane shfl row reductions; P C->A layout via per-wave
// LDS; O += P @ V with 4 wmma.

#define KLD 72  // 64 + 8 pad (halves)
#define VLD 40  // 32 + 8 pad
#define PLD 40

__global__ __launch_bounds__(256)
void attn_wmma_kernel(const half_t* __restrict__ q_h,          // [b*TQ][1024]
                      const half_t* __restrict__ k_h, int ldk, // [b*TKV][ldk]
                      const half_t* __restrict__ v_t,          // [(b*H+h)*64+d][TKV]
                      const int* __restrict__ attn_mask,
                      const int* __restrict__ q_mask,
                      const int* __restrict__ kv_mask,
                      half_t* __restrict__ o_h) {
  __shared__ __attribute__((aligned(16))) half_t Klds[32 * KLD];   // [kv][d]
  __shared__ __attribute__((aligned(16))) half_t Vlds[64 * VLD];   // [d][kv]
  __shared__ __attribute__((aligned(16))) half_t Plds[8 * 16 * PLD];
  const int tid  = threadIdx.x;
  const int lane = tid & 31;
  const int wave = tid >> 5;
  const int bx = blockIdx.x;
  const int qb = bx & 15;
  const int h  = (bx >> 4) & (HEADS_ - 1);
  const int b  = bx >> 8;
  const int qrow0 = qb * 128 + wave * 16;
  const int hi = lane >> 4, nn = lane & 15;

  const half_t* qbase = q_h + (size_t)(b * TQ_ + qrow0) * DIM_ + h * DHEAD_;
  const v16h aq0 = load_frag_a(qbase, DIM_);
  const v16h aq1 = load_frag_a(qbase + 32, DIM_);

  // per-thread staging coordinates
  const int kn = tid >> 3, kdc = (tid & 7) * 8;   // K tile: 32 rows x 64 d
  const int dv = tid >> 2, jc  = (tid & 3) * 8;   // V tile: 64 d x 32 kv
  const size_t kelem0 = (size_t)(b * TKV_ + kn) * ldk + h * DHEAD_ + kdc;
  const size_t velem0 = (size_t)((b * HEADS_ + h) * DHEAD_ + dv) * TKV_ + jc;

  float row_max[8], row_sum[8];
  v8f oacc[4];
  v8f zero = {};
#pragma unroll
  for (int i = 0; i < 8; ++i) { row_max[i] = -__builtin_inff(); row_sum[i] = 0.f; }
#pragma unroll
  for (int t = 0; t < 4; ++t) oacc[t] = zero;

  half_t* pl = &Plds[wave * 16 * PLD];

  for (int j0 = 0; j0 < TKV_; j0 += 32) {
    __syncthreads();  // previous iteration's K/V reads complete
#if USE_ASYNC_LDS
    async_g2l_b128(&Klds[kn * KLD + kdc], k_h,
                   (unsigned)((kelem0 + (size_t)j0 * ldk) * 2));
    async_g2l_b128(&Vlds[dv * VLD + jc], v_t,
                   (unsigned)((velem0 + (size_t)j0) * 2));
    async_wait0();
#else
    {
      uint4 kv4 = *reinterpret_cast<const uint4*>(k_h + kelem0 + (size_t)j0 * ldk);
      *reinterpret_cast<uint4*>(&Klds[kn * KLD + kdc]) = kv4;
      uint4 vv4 = *reinterpret_cast<const uint4*>(v_t + velem0 + (size_t)j0);
      *reinterpret_cast<uint4*>(&Vlds[dv * VLD + jc]) = vv4;
    }
#endif
    __builtin_prefetch(k_h + kelem0 + (size_t)(j0 + 32) * ldk, 0, 0);
    __builtin_prefetch(v_t + velem0 + (size_t)(j0 + 32), 0, 0);
    __syncthreads();  // tiles visible to all waves

    // ---- sim = Q @ K^T for 32 keys (two 16-col tiles) ----
    v8f s0 = zero, s1 = zero;
    s0 = wmma16(aq0, load_frag_b(&Klds[0 * KLD + 0], KLD), s0);
    s0 = wmma16(aq1, load_frag_b(&Klds[0 * KLD + 32], KLD), s0);
    s1 = wmma16(aq0, load_frag_b(&Klds[16 * KLD + 0], KLD), s1);
    s1 = wmma16(aq1, load_frag_b(&Klds[16 * KLD + 32], KLD), s1);

    const int col0 = j0 + nn, col1 = j0 + 16 + nn;
    const int kvm0 = kv_mask[b * TKV_ + col0];
    const int kvm1 = kv_mask[b * TKV_ + col1];

    // ---- masked online softmax (row vg across this 16-lane group) ----
#pragma unroll
    for (int vg = 0; vg < 8; ++vg) {
      const int row = qrow0 + vg + 8 * hi;
      const int* amrow = attn_mask + (size_t)(b * TQ_ + row) * TKV_;
      float e0 = (kvm0 && amrow[col0]) ? s0[vg] : -__builtin_inff();
      float e1 = (kvm1 && amrow[col1]) ? s1[vg] : -__builtin_inff();
      float m = fmaxf(e0, e1);
#pragma unroll
      for (int msk = 8; msk >= 1; msk >>= 1)
        m = fmaxf(m, __shfl_xor(m, msk, 32));
      const float nm = fmaxf(row_max[vg], m);
      const bool dead = !(nm > -1.0e30f);
      const float alpha = dead ? 1.0f : __expf(row_max[vg] - nm);
      const float p0 = dead ? 0.0f : __expf(e0 - nm);
      const float p1 = dead ? 0.0f : __expf(e1 - nm);
      float psum = p0 + p1;
#pragma unroll
      for (int msk = 8; msk >= 1; msk >>= 1)
        psum += __shfl_xor(psum, msk, 32);
      row_sum[vg] = row_sum[vg] * alpha + psum;
      row_max[vg] = nm;
#pragma unroll
      for (int t = 0; t < 4; ++t) oacc[t][vg] *= alpha;
      const int mrow = vg + 8 * hi;
      pl[mrow * PLD + nn]      = (half_t)p0;
      pl[mrow * PLD + 16 + nn] = (half_t)p1;
    }
    asm volatile("s_wait_dscnt 0x0" ::: "memory");  // wave-private C->A round-trip
    const v16h pa = load_frag_a(pl, PLD);

    // ---- O += P(16x32) @ V(32x64) ----
#pragma unroll
    for (int t = 0; t < 4; ++t)
      oacc[t] = wmma16(pa, load_frag_b(&Vlds[(t * 16) * VLD], VLD), oacc[t]);
  }

  // ---- finalize: 1/sum (0 when row dead or q_mask off), store f16 ----
#pragma unroll
  for (int vg = 0; vg < 8; ++vg) {
    const int row = qrow0 + vg + 8 * hi;
    float inv = 0.0f;
    if (row_sum[vg] > 0.0f && q_mask[b * TQ_ + row]) inv = 1.0f / row_sum[vg];
#pragma unroll
    for (int t = 0; t < 4; ++t) {
      o_h[(size_t)(b * TQ_ + row) * DIM_ + h * DHEAD_ + t * 16 + nn] =
          (half_t)(oacc[t][vg] * inv);
    }
  }
}

// ------------------------------ launcher -----------------------------------

extern "C" void kernel_launch(void* const* d_in, const int* in_sizes, int n_in,
                              void* d_out, int out_size, void* d_ws, size_t ws_size,
                              hipStream_t stream) {
  (void)in_sizes; (void)n_in; (void)out_size; (void)ws_size;
  const float* q_in   = (const float*)d_in[0];
  const float* kv_in  = (const float*)d_in[1];
  const int* attn_mask = (const int*)d_in[2];
  const int* q_mask    = (const int*)d_in[3];
  const int* kv_mask   = (const int*)d_in[4];
  const float* gamma_q = (const float*)d_in[5];
  const float* beta_q  = (const float*)d_in[6];
  const float* gamma_kv = (const float*)d_in[7];
  const float* beta_kv  = (const float*)d_in[8];
  const float* Wq   = (const float*)d_in[9];
  const float* Wkv  = (const float*)d_in[10];
  const float* Wout = (const float*)d_in[11];
  const float* attn_gate = (const float*)d_in[12];
  const float* ff_gamma = (const float*)d_in[13];
  const float* ff_beta  = (const float*)d_in[14];
  const float* W1 = (const float*)d_in[15];
  const float* W2 = (const float*)d_in[16];
  const float* ff_gate = (const float*)d_in[17];
  float* out = (float*)d_out;

  size_t off = 0;
  char* base = (char*)d_ws;
  auto carve = [&](size_t bytes) {
    void* p = base + off;
    off += (bytes + 255) & ~(size_t)255;
    return p;
  };
  half_t* qn_h  = (half_t*)carve((size_t)MROWS_ * DIM_ * 2);
  half_t* kvn_h = (half_t*)carve((size_t)MROWS_ * DIM_ * 2);
  half_t* WqT   = (half_t*)carve((size_t)DIM_ * DIM_ * 2);     // [N][K]
  half_t* WkvT  = (half_t*)carve((size_t)DIM_ * 2048 * 2);
  half_t* WoutT = (half_t*)carve((size_t)DIM_ * DIM_ * 2);
  half_t* W1T   = (half_t*)carve((size_t)DIM_ * FFDIM_ * 2);
  half_t* W2T   = (half_t*)carve((size_t)FFDIM_ * DIM_ * 2);
  half_t* q_h   = (half_t*)carve((size_t)MROWS_ * DIM_ * 2);
  half_t* kv_h  = (half_t*)carve((size_t)MROWS_ * 2048 * 2);   // [k | v]
  half_t* vt_h  = (half_t*)carve((size_t)MROWS_ * DIM_ * 2);
  half_t* o_h   = (half_t*)carve((size_t)MROWS_ * DIM_ * 2);
  float*  x_f   = (float*)carve((size_t)MROWS_ * DIM_ * 4);
  half_t* xn_h  = (half_t*)carve((size_t)MROWS_ * DIM_ * 2);
  half_t* h1_h  = (half_t*)carve((size_t)MROWS_ * FFDIM_ * 2);

  // 1) weight cast+transpose f32[K][N] -> f16[N][K]
  auto castT = [&](const float* src, half_t* dst, int R, int C) {
    int n = R * C;
    cast_transpose_kernel<<<(n + 255) / 256, 256, 0, stream>>>(src, dst, R, C);
  };
  castT(Wq,   WqT,   DIM_, DIM_);
  castT(Wkv,  WkvT,  DIM_, 2048);
  castT(Wout, WoutT, DIM_, DIM_);
  castT(W1,   W1T,   DIM_, FFDIM_);
  castT(W2,   W2T,   FFDIM_, DIM_);

  // 2) layernorms (fused cast)
  layernorm_cast_kernel<<<MROWS_, 256, 0, stream>>>(q_in, gamma_q, beta_q, qn_h);
  layernorm_cast_kernel<<<MROWS_, 256, 0, stream>>>(kv_in, gamma_kv, beta_kv, kvn_h);

  // 3) projections (1/sqrt(64) folded into q)
  gemm_wmma_kernel<<<dim3(DIM_ / 128, MROWS_ / 128), 256, 0, stream>>>(
      qn_h, WqT, DIM_, DIM_, DIM_, /*mode=*/0, 0.125f,
      nullptr, nullptr, q_h, nullptr);
  gemm_wmma_kernel<<<dim3(2048 / 128, MROWS_ / 128), 256, 0, stream>>>(
      kvn_h, WkvT, DIM_, DIM_, 2048, /*mode=*/0, 1.0f,
      nullptr, nullptr, kv_h, nullptr);

  // 4) V transpose -> [b,h,d,kv]
  {
    int n = B_ * TKV_ * HEADS_ * DHEAD_;
    transpose_v_kernel<<<(n + 255) / 256, 256, 0, stream>>>(kv_h, vt_h);
  }

  // 5) flash attention
  attn_wmma_kernel<<<B_ * HEADS_ * (TQ_ / 128), 256, 0, stream>>>(
      q_h, kv_h, 2048, vt_h, attn_mask, q_mask, kv_mask, o_h);

  // 6) output projection + attn gate + residual -> x_f
  gemm_wmma_kernel<<<dim3(DIM_ / 128, MROWS_ / 128), 256, 0, stream>>>(
      o_h, WoutT, DIM_, DIM_, DIM_, /*mode=*/2, 1.0f,
      attn_gate, q_in, nullptr, x_f);

  // 7) FFN
  layernorm_cast_kernel<<<MROWS_, 256, 0, stream>>>(x_f, ff_gamma, ff_beta, xn_h);
  gemm_wmma_kernel<<<dim3(FFDIM_ / 128, MROWS_ / 128), 256, 0, stream>>>(
      xn_h, W1T, DIM_, DIM_, FFDIM_, /*mode=*/1, 1.0f,
      nullptr, nullptr, h1_h, nullptr);
  gemm_wmma_kernel<<<dim3(DIM_ / 128, MROWS_ / 128), 256, 0, stream>>>(
      h1_h, W2T, FFDIM_, FFDIM_, DIM_, /*mode=*/2, 1.0f,
      ff_gate, x_f, nullptr, out);
}
